// CGNN_88038239634099
// MI455X (gfx1250) — matrix operand order repaired and checked by reading, not compiled, not run on hardware
//
#include <hip/hip_runtime.h>
#include <hip/hip_bf16.h>

// ---------------- problem constants ----------------
#define BSZ   4096
#define NN    64
#define HD    128
#define LAYERS 3
#define EDG   1024
#define OUTC  2
#define KTOT  (NN * HD)      // 8192 classifier K
#define NTH   (HD / 16)      // 8 N-tiles across H

#define BLK   256            // 8 wave32s
#define NB    2              // batches per GNN workgroup
#define RT    (NB * NN)      // 128 rows of h per workgroup
#define LDH   132            // padded leading dim (floats) -> 4-bank shift per row
#define LDT   136            // padded bf16 leading dim for transposed h_new
#define CMT   64             // classifier M-tile (batch rows per WG)
#define KC    128            // classifier K chunk
#define LDA   136            // padded bf16 leading dim for classifier A tile

// dynamic LDS sizes
#define GNN_LDS  (RT*LDH*4 /*h*/ + HD*LDT*2 /*h_newT bf16*/ + HD*HD*2 /*W frag*/ + NN*NN*2 /*adj frag*/ + HD*4 /*bias*/)
#define CLS_LDS  (CMT*LDA*2 /*A bf16*/ + KC*HD*2 /*W frag*/ + CMT*HD*4 /*hidden f32*/)

// ws layout (bytes)
#define WS_ADJF   0                        // 64*64 f32      = 16384
#define WS_ADJSW  16384                    // 64*64 bf16     = 8192   (A-fragment layout)
#define WS_WSW    24576                    // 3*128*128 bf16 = 98304  (B-fragment layout)
#define WS_WC1SW  122880                   // 8192*128 bf16  = 2097152 (B-fragment layout)
#define WS_HBUF   2220032                  // 4096*8192 bf16 = 67108864

typedef __attribute__((ext_vector_type(16))) __bf16 v16bf;
typedef __attribute__((ext_vector_type(8)))  float  v8f;

__device__ __forceinline__ v8f wmma_bf16(v16bf a, v16bf b, v8f c) {
  return __builtin_amdgcn_wmma_f32_16x16x32_bf16(false, a, false, b, (short)0, c, false, false);
}

// ---------- fragment-native storage: [tile][half(2)][lane(32)][8 x bf16] ----------
__device__ __forceinline__ v16bf load_frag_sw(const __bf16* __restrict__ base, int tile) {
  const int lane = threadIdx.x & 31;
  const uint4* p = (const uint4*)(base + tile * 512);
  union { uint4 u[2]; v16bf v; } c;
  c.u[0] = p[lane];
  c.u[1] = p[lane + 32];
  return c.v;
}

// decode flat fragment-storage index -> (tile, lane, e)
__device__ __forceinline__ void decode_frag(int i, int& t, int& lane, int& e) {
  e    = (((i >> 8) & 1) << 3) | (i & 7);
  lane = (i >> 3) & 31;
  t    = i >> 9;
}

// ---- A fragment (16x32 bf16) from row-major f32 LDS tile: 4x ds_load_b128 + pk-cvt ----
__device__ __forceinline__ v16bf a_frag_f32_vec(const float* __restrict__ s, int ld) {
  const int lane = threadIdx.x & 31;
  const float* row = s + (lane & 15) * ld + ((lane >> 4) << 3);
  float4 f0 = *(const float4*)(row + 0);    // K = hi+0..3   -> e 0..3
  float4 f1 = *(const float4*)(row + 4);    // K = hi+4..7   -> e 4..7
  float4 f2 = *(const float4*)(row + 16);   // K = hi+16..19 -> e 8..11
  float4 f3 = *(const float4*)(row + 20);   // K = hi+20..23 -> e 12..15
  v16bf a;
  a[0]=(__bf16)f0.x;  a[1]=(__bf16)f0.y;  a[2]=(__bf16)f0.z;  a[3]=(__bf16)f0.w;
  a[4]=(__bf16)f1.x;  a[5]=(__bf16)f1.y;  a[6]=(__bf16)f1.z;  a[7]=(__bf16)f1.w;
  a[8]=(__bf16)f2.x;  a[9]=(__bf16)f2.y;  a[10]=(__bf16)f2.z; a[11]=(__bf16)f2.w;
  a[12]=(__bf16)f3.x; a[13]=(__bf16)f3.y; a[14]=(__bf16)f3.z; a[15]=(__bf16)f3.w;
  return a;
}

// ---- A fragment from row-major bf16 LDS tile: 2x ds_load_b128 ----
__device__ __forceinline__ v16bf a_frag_bf_vec(const __bf16* __restrict__ s, int ld) {
  const int lane = threadIdx.x & 31;
  const __bf16* row = s + (lane & 15) * ld + ((lane >> 4) << 3);
  union { uint4 u[2]; v16bf v; } c;
  c.u[0] = *(const uint4*)(row);        // K = hi..hi+7     -> e 0..7
  c.u[1] = *(const uint4*)(row + 16);   // K = hi+16..hi+23 -> e 8..15
  return c.v;
}

// ---- B fragment (32x16) from TRANSPOSED bf16 LDS (s[n*ldT + k]): 2x ds_load_b128 ----
__device__ __forceinline__ v16bf b_fragT_bf(const __bf16* __restrict__ s, int ldT) {
  const int lane = threadIdx.x & 31;
  const __bf16* col = s + (lane & 15) * ldT + ((lane >> 4) << 3);
  union { uint4 u[2]; v16bf v; } c;
  c.u[0] = *(const uint4*)(col);        // K = hi..hi+7     -> e 0..7
  c.u[1] = *(const uint4*)(col + 16);   // K = hi+16..hi+23 -> e 8..15
  return c.v;
}

// ---- pack D fragment rows (+bias) to 8 bf16 and store as one ds_store_b128 ----
__device__ __forceinline__ void store_d_bf8(__bf16* __restrict__ dst, const v8f& ac, float bias) {
  union { uint4 u; __bf16 h[8]; } st;
#pragma unroll
  for (int r = 0; r < 8; ++r) st.h[r] = (__bf16)(ac[r] + bias);
  *(uint4*)dst = st.u;
}

// ================= prep: zero adj, swizzle weights into B-fragment layout =================
__global__ void prep_kernel(const float* __restrict__ Wl, const float* __restrict__ Wc1f,
                            float* __restrict__ adjf, __bf16* __restrict__ Wsw,
                            __bf16* __restrict__ Wc1sw) {
  for (size_t ii = (size_t)blockIdx.x * blockDim.x + threadIdx.x;
       ii < (size_t)KTOT * HD; ii += (size_t)gridDim.x * blockDim.x) {
    int i = (int)ii;
    int t, lane, e; decode_frag(i, t, lane, e);
    { // Wc1: K x N = 8192 x 128, tile t = kt*NTH + nt
      int kt = t >> 3, nt = t & 7;
      int K = kt * 32 + ((lane >> 4) << 3) + (e & 7) + ((e & 8) << 1);
      int N_ = nt * 16 + (lane & 15);
      Wc1sw[i] = (__bf16)Wc1f[(size_t)K * HD + N_];
    }
    if (i < LAYERS * HD * HD) { // per-layer 128x128
      int l = i / (HD * HD), r = i % (HD * HD);
      int t2, lane2, e2; decode_frag(r, t2, lane2, e2);
      int kt = t2 >> 3, nt = t2 & 7;
      int K = kt * 32 + ((lane2 >> 4) << 3) + (e2 & 7) + ((e2 & 8) << 1);
      int N_ = nt * 16 + (lane2 & 15);
      Wsw[i] = (__bf16)Wl[l * HD * HD + K * HD + N_];
    }
    if (i < NN * NN) adjf[i] = 0.0f;
  }
}

// ================= densify edge list into Adj[dst][src] += w =================
__global__ void edge_kernel(const int* __restrict__ ei, const float* __restrict__ ea,
                            float* __restrict__ adjf) {
  int e = blockIdx.x * blockDim.x + threadIdx.x;
  if (e < EDG) {
    int s = ei[e];
    int d = ei[EDG + e];
    atomicAdd(&adjf[d * NN + s], ea[e]);
  }
}

// Adj -> A-fragment layout (Adj is the A operand of the message GEMM)
__global__ void adjsw_kernel(const float* __restrict__ adjf, __bf16* __restrict__ adjsw) {
  int i = blockIdx.x * blockDim.x + threadIdx.x;
  if (i < NN * NN) {
    int t, lane, e; decode_frag(i, t, lane, e);      // t = mt*2 + kt
    int mt = t >> 1, kt = t & 1;
    int K = kt * 32 + (e & 7) + ((e & 8) << 1) + ((lane >> 4) << 3);
    int M = mt * 16 + (lane & 15);
    adjsw[i] = (__bf16)adjf[M * NN + K];
  }
}

// ================= GNN: 3 layers fully LDS-resident, all GEMMs via WMMA =================
__global__ void __launch_bounds__(BLK, 1)
gnn_kernel(const float* __restrict__ x, const float* __restrict__ w_enc,
           const float* __restrict__ b_enc, const float* __restrict__ b_layers,
           const __bf16* __restrict__ Wsw, const __bf16* __restrict__ adjsw,
           __bf16* __restrict__ hbuf) {
  extern __shared__ char smem[];
  float*  sh_h   = (float*)smem;                    // RT x LDH fp32 (row-major, padded)
  __bf16* sh_hnT = (__bf16*)(sh_h + RT * LDH);      // HD cols x LDT bf16 (transposed h_new)
  __bf16* sh_w   = sh_hnT + HD * LDT;               // 16384 bf16, B-fragment layout
  __bf16* sh_adj = sh_w + HD * HD;                  // 4096 bf16, A-fragment layout
  float*  sh_b   = (float*)(sh_adj + NN * NN);      // HD bias

  const int tid  = threadIdx.x;
  const int wave = tid >> 5;
  const int lane = tid & 31;
  const int b0   = blockIdx.x * NB;

  // encode: h = x[...,None]*w_enc + b_enc  (h stays in LDS for all 3 layers)
  for (int idx = tid; idx < RT * HD; idx += BLK) {
    int row = idx >> 7, k = idx & (HD - 1);
    sh_h[row * LDH + k] = x[b0 * NN + row] * w_enc[k] + b_enc[k];
  }
  for (int idx = tid; idx < (NN * NN) / 8; idx += BLK)
    ((uint4*)sh_adj)[idx] = ((const uint4*)adjsw)[idx];

  for (int l = 0; l < LAYERS; ++l) {
    const uint4* wsrc = (const uint4*)(Wsw + l * HD * HD);
    for (int idx = tid; idx < (HD * HD) / 8; idx += BLK)
      ((uint4*)sh_w)[idx] = wsrc[idx];
    if (tid < HD) sh_b[tid] = b_layers[l * HD + tid];
    __syncthreads();

    // GEMM1: h_new = h(128x128) @ W(128x128) + b ; wave owns mt, 8 independent accumulators.
    // A fragment loaded/converted ONCE per K-step, 8 back-to-back independent WMMAs.
    {
      const int mt = wave;
      v8f acc[8] = {};
#pragma unroll
      for (int kt = 0; kt < 4; ++kt) {
        v16bf a = a_frag_f32_vec(sh_h + mt * 16 * LDH + kt * 32, LDH);
#pragma unroll
        for (int nt = 0; nt < 8; ++nt)
          acc[nt] = wmma_bf16(a, load_frag_sw(sh_w, kt * NTH + nt), acc[nt]);
      }
      const int row0 = mt * 16 + ((lane >> 4) << 3);
#pragma unroll
      for (int nt = 0; nt < 8; ++nt) {
        const int col = nt * 16 + (lane & 15);
        store_d_bf8(sh_hnT + col * LDT + row0, acc[nt], sh_b[nt * 16 + (lane & 15)]);
      }
    }
    __syncthreads();

    // GEMM2: messages_b = Adj(64x64) @ h_new_b(64x128); h = relu(h + messages)
    // wave owns nt; B fragments (bf16, contiguous) loaded once per (bb,kt); A shared across bb.
    {
      const int nt = wave;
      v8f acc[8] = {};                       // [bb*4 + mt]
#pragma unroll
      for (int kt = 0; kt < 2; ++kt) {
        v16bf bv0 = b_fragT_bf(sh_hnT + (nt * 16) * LDT + 0 * NN + kt * 32, LDT);
        v16bf bv1 = b_fragT_bf(sh_hnT + (nt * 16) * LDT + 1 * NN + kt * 32, LDT);
#pragma unroll
        for (int mt = 0; mt < 4; ++mt) {
          v16bf a = load_frag_sw(sh_adj, mt * 2 + kt);
          acc[mt]     = wmma_bf16(a, bv0, acc[mt]);
          acc[4 + mt] = wmma_bf16(a, bv1, acc[4 + mt]);
        }
      }
#pragma unroll
      for (int bb = 0; bb < NB; ++bb)
#pragma unroll
        for (int mt = 0; mt < 4; ++mt) {
          const v8f ac = acc[bb * 4 + mt];
          const int row0 = bb * NN + mt * 16 + ((lane >> 4) << 3);
          const int col  = nt * 16 + (lane & 15);
#pragma unroll
          for (int r = 0; r < 8; ++r) {
            const int idx = (row0 + r) * LDH + col;
            float v = sh_h[idx] + ac[r];
            sh_h[idx] = v > 0.0f ? v : 0.0f;
          }
        }
    }
    __syncthreads();
  }

  // export final h as bf16 (classifier A operand)
  for (int idx = tid; idx < RT * HD; idx += BLK) {
    int row = idx >> 7, k = idx & (HD - 1);
    hbuf[(size_t)b0 * KTOT + idx] = (__bf16)sh_h[row * LDH + k];
  }
}

// ================= classifier: (4096 x 8192) @ (8192 x 128) + relu, then (128 x 2) =================
__global__ void __launch_bounds__(BLK, 1)
classifier_kernel(const __bf16* __restrict__ hbuf, const __bf16* __restrict__ Wc1sw,
                  const float* __restrict__ bc1, const float* __restrict__ Wc2,
                  const float* __restrict__ bc2, float* __restrict__ out) {
  extern __shared__ char smem[];
  __bf16* sh_a   = (__bf16*)smem;                  // CMT x LDA bf16 (padded rows)
  __bf16* sh_wc  = sh_a + CMT * LDA;               // 16384 bf16, B-fragment layout
  float*  sh_hid = (float*)(sh_wc + KC * HD);      // CMT x HD fp32

  const int tid  = threadIdx.x;
  const int wave = tid >> 5;
  const int lane = tid & 31;
  const int m0   = blockIdx.x * CMT;

  const int mt  = wave >> 1;          // wave owns (mt, nt-group of 4)
  const int nts = (wave & 1) * 4;

  v8f acc[4] = {};
  for (int kc = 0; kc < KTOT; kc += KC) {
    const int ck = kc >> 7;                        // chunk index
    // stage A chunk (64 rows x 128 bf16, padded) — 16 uint4 per row
    for (int idx = tid; idx < CMT * (KC / 8); idx += BLK) {
      int r = idx >> 4, c8 = idx & 15;
      ((uint4*)(sh_a + r * LDA))[c8] =
          ((const uint4*)(hbuf + (size_t)(m0 + r) * KTOT + kc))[c8];
    }
    // stage W chunk: contiguous copy of pre-swizzled fragments (4 k-tiles x 8 n-tiles)
    {
      const uint4* src = (const uint4*)(Wc1sw + (size_t)ck * (KC * HD));
      uint4* dst = (uint4*)sh_wc;
      for (int idx = tid; idx < (KC * HD) / 8; idx += BLK) dst[idx] = src[idx];
    }
    if (kc + KC < KTOT)  // hint next weight chunk into cache (global_prefetch_b8)
      __builtin_prefetch(Wc1sw + (size_t)(ck + 1) * (KC * HD) + tid * 64, 0, 1);
    __syncthreads();

#pragma unroll
    for (int kk = 0; kk < 4; ++kk) {
      v16bf a = a_frag_bf_vec(sh_a + mt * 16 * LDA + kk * 32, LDA);
#pragma unroll
      for (int j = 0; j < 4; ++j)
        acc[j] = wmma_bf16(a, load_frag_sw(sh_wc, kk * NTH + nts + j), acc[j]);
    }
    __syncthreads();
  }

  // bias + relu -> hidden in LDS
  {
    const int row0 = mt * 16 + ((lane >> 4) << 3);
#pragma unroll
    for (int j = 0; j < 4; ++j) {
      const int nt = nts + j;
      const float bias = bc1[nt * 16 + (lane & 15)];
      const int col = nt * 16 + (lane & 15);
#pragma unroll
      for (int r = 0; r < 8; ++r) {
        float v = acc[j][r] + bias;
        sh_hid[(row0 + r) * HD + col] = v > 0.0f ? v : 0.0f;
      }
    }
  }
  __syncthreads();

  // logits = hidden @ Wc2 + bc2  (O=2: per-thread 128-dot)
  if (tid < CMT * OUTC) {
    const int row = tid >> 1, o = tid & 1;
    float s = bc2[o];
#pragma unroll 8
    for (int k = 0; k < HD; ++k) s += sh_hid[row * HD + k] * Wc2[k * OUTC + o];
    out[(size_t)(m0 + row) * OUTC + o] = s;
  }
}

// ================= launcher =================
extern "C" void kernel_launch(void* const* d_in, const int* in_sizes, int n_in,
                              void* d_out, int out_size, void* d_ws, size_t ws_size,
                              hipStream_t stream) {
  const float* x      = (const float*)d_in[0];
  const int*   ei     = (const int*)d_in[1];     // edge_index (2,E)
  const float* ea     = (const float*)d_in[2];   // edge_attr (E,1)
  const float* w_enc  = (const float*)d_in[3];
  const float* b_enc  = (const float*)d_in[4];
  const float* Wl     = (const float*)d_in[5];
  const float* bl     = (const float*)d_in[6];
  const float* Wc1    = (const float*)d_in[7];
  const float* bc1    = (const float*)d_in[8];
  const float* Wc2    = (const float*)d_in[9];
  const float* bc2    = (const float*)d_in[10];
  float* out = (float*)d_out;

  char* ws = (char*)d_ws;
  float*  adjf  = (float*)(ws + WS_ADJF);
  __bf16* adjsw = (__bf16*)(ws + WS_ADJSW);
  __bf16* Wsw   = (__bf16*)(ws + WS_WSW);
  __bf16* Wc1sw = (__bf16*)(ws + WS_WC1SW);
  __bf16* hbuf  = (__bf16*)(ws + WS_HBUF);

  (void)hipFuncSetAttribute(reinterpret_cast<const void*>(&gnn_kernel),
                            hipFuncAttributeMaxDynamicSharedMemorySize, GNN_LDS);
  (void)hipFuncSetAttribute(reinterpret_cast<const void*>(&classifier_kernel),
                            hipFuncAttributeMaxDynamicSharedMemorySize, CLS_LDS);

  prep_kernel<<<(KTOT * HD) / BLK, BLK, 0, stream>>>(Wl, Wc1, adjf, Wsw, Wc1sw);
  edge_kernel<<<(EDG + BLK - 1) / BLK, BLK, 0, stream>>>(ei, ea, adjf);
  adjsw_kernel<<<(NN * NN + BLK - 1) / BLK, BLK, 0, stream>>>(adjf, adjsw);
  gnn_kernel<<<BSZ / NB, BLK, GNN_LDS, stream>>>(x, w_enc, b_enc, bl, Wsw, adjsw, hbuf);
  classifier_kernel<<<BSZ / CMT, BLK, CLS_LDS, stream>>>(hbuf, Wc1sw, bc1, Wc2, bc2, out);
}
